// get_model_27771258536242
// MI455X (gfx1250) — matrix operand order repaired and checked by reading, not compiled, not run on hardware
//
#include <hip/hip_runtime.h>
#include <hip/hip_bf16.h>

// ============================================================================
// PointNet++ MSG part-seg forward for MI455X (gfx1250, wave32, WMMA f16).
//
// Round-2 changes (driven by disasm): all WMMA K-loops are now guard-free.
//  * Weights are pre-converted once to f16 [pad32(Cout), pad32(Cin)], zero
//    padded -> B fragments are two contiguous b128 loads.
//  * Dense-chain activations live in f16 [M, pad32(C)] buffers; GEMM epilogues
//    zero-fill pad columns so downstream K-tiles read zeros -> A fragments are
//    two contiguous b128 loads. No f32->f16 cvt, no EXEC juggling in the loop.
//  * Fused SA kernel stages the gathered group input into LDS as padded f16
//    once, then all 3 MLP layers run the same guard-free ds_load_b128 + wmma
//    loop; layer-3 tiles max-reduce into an LDS [C3] array (uint max trick).
//  * Dense GEMM computes a 16x64 strip per wave (4 N-tiles reuse A fragment).
// ============================================================================

typedef __attribute__((ext_vector_type(16))) _Float16 v16h;
typedef __attribute__((ext_vector_type(8)))  _Float16 v8h;
typedef __attribute__((ext_vector_type(8)))  float    v8f;

#define BN_RSQ 0.9999950000374997f   // rsqrt(1 + 1e-5)

static __device__ __forceinline__ v16h cat16(v8h lo, v8h hi)
{
    return __builtin_shufflevector(lo, hi, 0, 1, 2, 3, 4, 5, 6, 7,
                                           8, 9, 10, 11, 12, 13, 14, 15);
}

// ---------------------------------------------------------------------------
// Weight conversion: W f32 [Cout,Cin] -> f16 [CoutP, CinP], zero padded.
// ---------------------------------------------------------------------------
__global__ void wcvt_kernel(const float* __restrict__ W, _Float16* __restrict__ O,
                            int Cout, int Cin, int CinP, int total)
{
    int i = blockIdx.x * blockDim.x + threadIdx.x;
    if (i >= total) return;
    int r = i / CinP, c = i % CinP;
    O[i] = (r < Cout && c < Cin) ? (_Float16)W[(size_t)r * Cin + c] : (_Float16)0.f;
}

// ---------------------------------------------------------------------------
// Dense WMMA linear, f16 in / f16 out, fully padded (no guards in K loop).
//   X:[M,CinP] f16, W16:[CoutP,CinP] f16, Y:[M,CoutP] f16 (pad cols zeroed)
//   Requires: M % 16 == 0, CinP % 32 == 0, CoutP % 64 == 0.
//   flags: 1 = BN affine, 2 = ReLU
// ---------------------------------------------------------------------------
__global__ void __launch_bounds__(256) wmma_linear16_kernel(
    const _Float16* __restrict__ X, const _Float16* __restrict__ W16,
    const float* __restrict__ bias, const float* __restrict__ g,
    const float* __restrict__ be, _Float16* __restrict__ Y,
    int M, int CinP, int Cout, int CoutP, int flags)
{
    const int nTN = CoutP >> 6;                      // 64-wide strips
    const int wave = blockIdx.x * 8 + (threadIdx.x >> 5);
    if (wave >= (M >> 4) * nTN) return;
    const int tm = wave / nTN, tsn = (wave % nTN) << 2;   // first 16-col tile
    const int lane = threadIdx.x & 31, lr = lane & 15, hi = lane >> 4;

    const _Float16* xrow = X + (size_t)(tm * 16 + lr) * CinP;
    const _Float16* wrow0 = W16 + (size_t)((tsn + 0) * 16 + lr) * CinP;
    const _Float16* wrow1 = W16 + (size_t)((tsn + 1) * 16 + lr) * CinP;
    const _Float16* wrow2 = W16 + (size_t)((tsn + 2) * 16 + lr) * CinP;
    const _Float16* wrow3 = W16 + (size_t)((tsn + 3) * 16 + lr) * CinP;

    v8f acc0 = {}, acc1 = {}, acc2 = {}, acc3 = {};
    const int kt = CinP >> 5;
    for (int k0 = 0; k0 < kt; ++k0) {
        const int kb = k0 * 32;
        v16h a = cat16(*(const v8h*)(xrow + kb + hi * 16),
                       *(const v8h*)(xrow + kb + hi * 16 + 8));
        v16h b0 = cat16(*(const v8h*)(wrow0 + kb + hi * 16),
                        *(const v8h*)(wrow0 + kb + hi * 16 + 8));
        v16h b1 = cat16(*(const v8h*)(wrow1 + kb + hi * 16),
                        *(const v8h*)(wrow1 + kb + hi * 16 + 8));
        v16h b2 = cat16(*(const v8h*)(wrow2 + kb + hi * 16),
                        *(const v8h*)(wrow2 + kb + hi * 16 + 8));
        v16h b3 = cat16(*(const v8h*)(wrow3 + kb + hi * 16),
                        *(const v8h*)(wrow3 + kb + hi * 16 + 8));
        acc0 = __builtin_amdgcn_wmma_f32_16x16x32_f16(false, a, false, b0, (short)0, acc0, false, false);
        acc1 = __builtin_amdgcn_wmma_f32_16x16x32_f16(false, a, false, b1, (short)0, acc1, false, false);
        acc2 = __builtin_amdgcn_wmma_f32_16x16x32_f16(false, a, false, b2, (short)0, acc2, false, false);
        acc3 = __builtin_amdgcn_wmma_f32_16x16x32_f16(false, a, false, b3, (short)0, acc3, false, false);
    }

    v8f accs[4] = {acc0, acc1, acc2, acc3};
#pragma unroll
    for (int j = 0; j < 4; ++j) {
        const int col = (tsn + j) * 16 + lr;
        const bool valid = col < Cout;
        float bb = 0.f, sc = 1.f, sh = 0.f;
        if (valid) {
            bb = bias ? bias[col] : 0.f;
            if (flags & 1) { sc = g[col] * BN_RSQ; sh = be[col]; }
        }
#pragma unroll
        for (int r = 0; r < 8; ++r) {
            const int m = tm * 16 + hi * 8 + r;
            float v = valid ? (accs[j][r] + bb) * sc + sh : 0.f;
            if (flags & 2) v = fmaxf(v, 0.f);
            Y[(size_t)m * CoutP + col] = (_Float16)v;
        }
    }
}

// ---------------------------------------------------------------------------
// Fused SA group kernel: stage gather in LDS -> 3x WMMA MLP -> max-pool.
// ---------------------------------------------------------------------------
struct SAParams {
    const float* points;   // [B,P,Cpts] f32
    const float* coords;   // [B,P,3]    f32
    const float* new_xyz;  // [B,S,3]    f32
    const int*   gi;       // [B,S,ns]
    int S, P, Cpts, ns;
    int Cin1P;             // pad32(Cpts+3)
    const _Float16 *W1; const float *b1, *g1, *be1; int C1;        // C1 % 32 == 0
    const _Float16 *W2; const float *b2, *g2, *be2; int C2, C2P;   // C2P % 32 == 0
    const _Float16 *W3; const float *b3, *g3, *be3; int C3;        // C3 % 16 == 0
    int actBhalfs;         // max(ns*Cin1P, ns*C2P)
    float* out; int outC, outOff;   // out[B,S,outC] f32
};

__device__ __forceinline__ void sa_gemm(
    const _Float16* __restrict__ actIn, int CinP,
    const _Float16* __restrict__ W16,
    const float* __restrict__ bias, const float* __restrict__ g,
    const float* __restrict__ be,
    int Cout, int CoutP,
    _Float16* __restrict__ actOut, unsigned* __restrict__ omax, int ns)
{
    const int lane = threadIdx.x & 31, lr = lane & 15, hi = lane >> 4;
    const int wave = threadIdx.x >> 5, nw = blockDim.x >> 5;
    const int nTM = ns >> 4, nTN = CoutP >> 4;
    const int kt = CinP >> 5;
    for (int t = wave; t < nTM * nTN; t += nw) {
        const int tm = t / nTN, tn = t % nTN;
        const _Float16* xrow = actIn + (size_t)(tm * 16 + lr) * CinP;
        const _Float16* wrow = W16 + (size_t)(tn * 16 + lr) * CinP;
        v8f acc = {};
        for (int k0 = 0; k0 < kt; ++k0) {
            const int kb = k0 * 32;
            v16h a = cat16(*(const v8h*)(xrow + kb + hi * 16),
                           *(const v8h*)(xrow + kb + hi * 16 + 8));
            v16h bf = cat16(*(const v8h*)(wrow + kb + hi * 16),
                            *(const v8h*)(wrow + kb + hi * 16 + 8));
            acc = __builtin_amdgcn_wmma_f32_16x16x32_f16(false, a, false, bf,
                                                         (short)0, acc, false, false);
        }
        const int col = tn * 16 + lr;
        const bool valid = col < Cout;
        float bb = 0.f, sc = 1.f, sh = 0.f;
        if (valid) { bb = bias[col]; sc = g[col] * BN_RSQ; sh = be[col]; }
#pragma unroll
        for (int r = 0; r < 8; ++r) {
            const int m = tm * 16 + hi * 8 + r;
            float v = valid ? fmaxf((acc[r] + bb) * sc + sh, 0.f) : 0.f;
            if (actOut) actOut[(size_t)m * CoutP + col] = (_Float16)v;
            else if (valid) atomicMax(&omax[col], __float_as_uint(v));  // v>=0
        }
    }
}

__global__ void __launch_bounds__(256) sa_fused_kernel(SAParams q)
{
    extern __shared__ char smem[];
    const int ns = q.ns;
    _Float16* actA = (_Float16*)smem;                       // ns * C1
    _Float16* actB = actA + (size_t)ns * q.C1;              // staging / ns * C2P
    unsigned* omax = (unsigned*)(actB + q.actBhalfs);       // C3
    int*      gis  = (int*)(omax + q.C3);                   // ns

    const int b = blockIdx.x / q.S;
    const int* gi = q.gi + (size_t)blockIdx.x * ns;
    for (int i = threadIdx.x; i < ns;   i += blockDim.x) gis[i]  = gi[i];
    for (int i = threadIdx.x; i < q.C3; i += blockDim.x) omax[i] = 0u;
    const float* center = q.new_xyz + (size_t)blockIdx.x * 3;
    const float* ptsB = q.points + (size_t)b * q.P * q.Cpts;
    const float* crdB = q.coords + (size_t)b * q.P * 3;
    __syncthreads();

    // --- stage gathered layer-1 input into LDS (padded f16), once
    {
        const int wave = threadIdx.x >> 5, lane = threadIdx.x & 31;
        const int nw = blockDim.x >> 5;
        for (int r = wave; r < ns; r += nw) {
            const int p = gis[r];
            const float* pr = ptsB + (size_t)p * q.Cpts;
            const float* cr = crdB + (size_t)p * 3;
            _Float16* dst = actB + (size_t)r * q.Cin1P;
            for (int c = lane; c < q.Cin1P; c += 32) {
                float v = 0.f;
                if (c < q.Cpts)            v = pr[c];
                else if (c < q.Cpts + 3)   v = cr[c - q.Cpts] - center[c - q.Cpts];
                dst[c] = (_Float16)v;
            }
        }
    }
    __syncthreads();
    sa_gemm(actB, q.Cin1P, q.W1, q.b1, q.g1, q.be1, q.C1, q.C1, actA, nullptr, ns);
    __syncthreads();
    sa_gemm(actA, q.C1, q.W2, q.b2, q.g2, q.be2, q.C2, q.C2P, actB, nullptr, ns);
    __syncthreads();
    sa_gemm(actB, q.C2P, q.W3, q.b3, q.g3, q.be3, q.C3, q.C3, nullptr, omax, ns);
    __syncthreads();

    float* outp = q.out + (size_t)blockIdx.x * q.outC + q.outOff;
    for (int c = threadIdx.x; c < q.C3; c += blockDim.x)
        outp[c] = __uint_as_float(omax[c]);
}

// ---------------------------------------------------------------------------
// Farthest point sampling: one block per batch; dist array in LDS.
// ---------------------------------------------------------------------------
__global__ void fps_kernel(const float* __restrict__ xyz, int P, int npoint,
                           int* __restrict__ out)
{
    extern __shared__ float smf[];
    float* dist = smf;
    float* rv = smf + P;
    int*   ri = (int*)(rv + blockDim.x);
    const int b = blockIdx.x;
    const float* X = xyz + (size_t)b * P * 3;
    for (int i = threadIdx.x; i < P; i += blockDim.x) dist[i] = 1e10f;
    __syncthreads();
    int far = 0;
    for (int it = 0; it < npoint; ++it) {
        if (threadIdx.x == 0) out[b * npoint + it] = far;
        const float cx = X[far * 3], cy = X[far * 3 + 1], cz = X[far * 3 + 2];
        float best = -1.f; int bi = 0;
        for (int i = threadIdx.x; i < P; i += blockDim.x) {
            float dx = X[i * 3] - cx, dy = X[i * 3 + 1] - cy, dz = X[i * 3 + 2] - cz;
            float nd = fminf(dist[i], dx * dx + dy * dy + dz * dz);
            dist[i] = nd;
            if (nd > best) { best = nd; bi = i; }
        }
        rv[threadIdx.x] = best; ri[threadIdx.x] = bi;
        __syncthreads();
        for (int off = blockDim.x >> 1; off > 0; off >>= 1) {
            if (threadIdx.x < off) {
                float ov = rv[threadIdx.x + off]; int oi = ri[threadIdx.x + off];
                if (ov > rv[threadIdx.x] ||
                    (ov == rv[threadIdx.x] && oi < ri[threadIdx.x])) {
                    rv[threadIdx.x] = ov; ri[threadIdx.x] = oi;
                }
            }
            __syncthreads();
        }
        far = ri[0];
        __syncthreads();
    }
}

// ---------------------------------------------------------------------------
// Utility kernels
// ---------------------------------------------------------------------------
__global__ void transpose_in_kernel(const float* __restrict__ xyz,
                                    float* __restrict__ xt,
                                    float* __restrict__ coords,
                                    int B, int C, int N)
{
    int i = blockIdx.x * blockDim.x + threadIdx.x;
    if (i >= B * N * C) return;
    int c = i % C; int n = (i / C) % N; int b = i / (C * N);
    float v = xyz[((size_t)b * C + c) * N + n];
    xt[((size_t)b * N + n) * C + c] = v;
    if (c < 3) coords[((size_t)b * N + n) * 3 + c] = v;
}

__global__ void gather3_kernel(const float* __restrict__ src,
                               const int* __restrict__ idx,
                               float* __restrict__ dst, int P, int S, int total)
{
    int i = blockIdx.x * blockDim.x + threadIdx.x;
    if (i >= total) return;
    int c = i % 3; int q = i / 3; int b = q / S;
    dst[i] = src[((size_t)b * P + idx[q]) * 3 + c];
}

__global__ void query_ball_kernel(const float* __restrict__ xyz,
                                  const float* __restrict__ new_xyz,
                                  float r2, int nsample, int P, int S,
                                  int* __restrict__ gi, int total)
{
    int q = blockIdx.x * blockDim.x + threadIdx.x;
    if (q >= total) return;
    int b = q / S;
    const float cx = new_xyz[q * 3], cy = new_xyz[q * 3 + 1], cz = new_xyz[q * 3 + 2];
    int* out = gi + (size_t)q * nsample;
    int cnt = 0, first = 0; bool hf = false;
    for (int p = 0; p < P && cnt < nsample; ++p) {
        const float* xp = xyz + ((size_t)b * P + p) * 3;
        float dx = xp[0] - cx, dy = xp[1] - cy, dz = xp[2] - cz;
        if (dx * dx + dy * dy + dz * dz <= r2) {
            if (!hf) { first = p; hf = true; }
            out[cnt++] = p;
        }
    }
    for (; cnt < nsample; ++cnt) out[cnt] = first;
}

__global__ void knn3_kernel(const float* __restrict__ xyz1,
                            const float* __restrict__ xyz2,
                            int N1, int N2, int* __restrict__ idx,
                            float* __restrict__ w, int total)
{
    int q = blockIdx.x * blockDim.x + threadIdx.x;
    if (q >= total) return;
    int b = q / N1;
    const float px = xyz1[q * 3], py = xyz1[q * 3 + 1], pz = xyz1[q * 3 + 2];
    float d0 = 1e30f, d1 = 1e30f, d2 = 1e30f; int i0 = 0, i1 = 0, i2 = 0;
    for (int p = 0; p < N2; ++p) {
        const float* xp = xyz2 + ((size_t)b * N2 + p) * 3;
        float dx = xp[0] - px, dy = xp[1] - py, dz = xp[2] - pz;
        float d = dx * dx + dy * dy + dz * dz;
        if (d < d0)      { d2 = d1; i2 = i1; d1 = d0; i1 = i0; d0 = d; i0 = p; }
        else if (d < d1) { d2 = d1; i2 = i1; d1 = d; i1 = p; }
        else if (d < d2) { d2 = d; i2 = p; }
    }
    float w0 = 1.f / (fmaxf(d0, 0.f) + 1e-8f);
    float w1 = 1.f / (fmaxf(d1, 0.f) + 1e-8f);
    float w2 = 1.f / (fmaxf(d2, 0.f) + 1e-8f);
    float ws = w0 + w1 + w2;
    idx[q * 3] = i0; idx[q * 3 + 1] = i1; idx[q * 3 + 2] = i2;
    w[q * 3] = w0 / ws; w[q * 3 + 1] = w1 / ws; w[q * 3 + 2] = w2 / ws;
}

__global__ void max_over_points16_kernel(const _Float16* __restrict__ in,
                                         float* __restrict__ out,
                                         int S, int C, int CP, int total)
{
    int i = blockIdx.x * blockDim.x + threadIdx.x;
    if (i >= total) return;
    int b = i / C, c = i % C;
    float m = -1e30f;
    for (int s = 0; s < S; ++s)
        m = fmaxf(m, (float)in[((size_t)b * S + s) * CP + c]);
    out[i] = m;
}

// concat builders (write padded f16) ----------------------------------------
__global__ void sa3_build_kernel(const float* __restrict__ l2xyz,
                                 const float* __restrict__ l2pts,
                                 _Float16* __restrict__ out, int total)  // [2048,544]
{
    int i = blockIdx.x * blockDim.x + threadIdx.x;
    if (i >= total) return;
    int c = i % 544, r = i / 544;
    float v = 0.f;
    if (c < 3)        v = l2xyz[r * 3 + c];
    else if (c < 515) v = l2pts[(size_t)r * 512 + (c - 3)];
    out[i] = (_Float16)v;
}

__global__ void fp3_build_kernel(const float* __restrict__ l2pts,
                                 const float* __restrict__ l3,
                                 _Float16* __restrict__ out, int total)  // [2048,1536]
{
    int i = blockIdx.x * blockDim.x + threadIdx.x;
    if (i >= total) return;
    int c = i % 1536, r = i / 1536, b = r >> 7;
    float v = (c < 512) ? l2pts[(size_t)r * 512 + c] : l3[(size_t)b * 1024 + (c - 512)];
    out[i] = (_Float16)v;
}

__global__ void fp2_build_kernel(const float* __restrict__ l1pts,
                                 const _Float16* __restrict__ l2new,  // [2048,256]
                                 const int* __restrict__ idx,
                                 const float* __restrict__ wgt,
                                 _Float16* __restrict__ out, int total)  // [8192,576]
{
    int i = blockIdx.x * blockDim.x + threadIdx.x;
    if (i >= total) return;
    int c = i % 576, r = i / 576, b = r >> 9;
    float v;
    if (c < 320) v = l1pts[(size_t)r * 320 + c];
    else {
        int cc = c - 320;
        v = 0.f;
#pragma unroll
        for (int j = 0; j < 3; ++j)
            v += wgt[r * 3 + j] *
                 (float)l2new[((size_t)b * 128 + idx[r * 3 + j]) * 256 + cc];
    }
    out[i] = (_Float16)v;
}

__global__ void fp1_build_kernel(const float* __restrict__ cls,
                                 const float* __restrict__ coords,
                                 const float* __restrict__ xt,
                                 const _Float16* __restrict__ l1new,  // [8192,128]
                                 const int* __restrict__ idx,
                                 const float* __restrict__ wgt,
                                 _Float16* __restrict__ out, int total)  // [32768,160]
{
    int i = blockIdx.x * blockDim.x + threadIdx.x;
    if (i >= total) return;
    int c = i % 160, r = i / 160, b = r >> 11;
    float v = 0.f;
    if (c < 16)       v = cls[b * 16 + c];
    else if (c < 19)  v = coords[(size_t)r * 3 + (c - 16)];
    else if (c < 25)  v = xt[(size_t)r * 6 + (c - 19)];
    else if (c < 153) {
        int cc = c - 25;
#pragma unroll
        for (int j = 0; j < 3; ++j)
            v += wgt[r * 3 + j] *
                 (float)l1new[((size_t)b * 512 + idx[r * 3 + j]) * 128 + cc];
    }
    out[i] = (_Float16)v;
}

__global__ void logsoftmax50_kernel(const _Float16* __restrict__ x,  // [rows,64]
                                    float* __restrict__ out, int rows)
{
    int r = blockIdx.x * blockDim.x + threadIdx.x;
    if (r >= rows) return;
    const _Float16* p = x + (size_t)r * 64;
    float m = (float)p[0];
    for (int i = 1; i < 50; ++i) m = fmaxf(m, (float)p[i]);
    float s = 0.f;
    for (int i = 0; i < 50; ++i) s += __expf((float)p[i] - m);
    float ls = logf(s);
    float* o = out + (size_t)r * 50;
    for (int i = 0; i < 50; ++i) o[i] = (float)p[i] - m - ls;
}

__global__ void copy_kernel(const float* __restrict__ src,
                            float* __restrict__ dst, int n)
{
    int i = blockIdx.x * blockDim.x + threadIdx.x;
    if (i < n) dst[i] = src[i];
}

// ===========================================================================
// Host-side orchestration
// ===========================================================================
struct Lyr { const float *W, *b, *g, *be; };

static inline int ceil_div(int a, int b) { return (a + b - 1) / b; }
static inline int pad32(int c) { return (c + 31) & ~31; }

static inline void launch_linear16(hipStream_t st, const _Float16* X,
                                   const _Float16* W16, const Lyr& L,
                                   _Float16* Y, int M, int Cin, int Cout, int flags)
{
    int CinP = pad32(Cin), CoutP = pad32(Cout);
    if (CoutP < 64) CoutP = 64;     // strip kernel needs CoutP % 64 == 0
    int tiles = (M / 16) * (CoutP / 64);
    wmma_linear16_kernel<<<ceil_div(tiles, 8), 256, 0, st>>>(
        X, W16, L.b, L.g, L.be, Y, M, CinP, Cout, CoutP, flags);
}

extern "C" void kernel_launch(void* const* d_in, const int* in_sizes, int n_in,
                              void* d_out, int out_size, void* d_ws, size_t ws_size,
                              hipStream_t stream)
{
    (void)in_sizes; (void)n_in; (void)out_size; (void)ws_size;
    const int B = 16, N = 2048;
    auto F = [&](int i) { return (const float*)d_in[i]; };

    // ---- parameter layout (recursive insertion order of setup_inputs dict)
    const float* xyz = F(0);
    const float* cls = F(1);
    int t = 2;
    auto nextL = [&](Lyr& L) { L.W = F(t); L.b = F(t + 1); L.g = F(t + 2); L.be = F(t + 3); t += 4; };
    Lyr sa1L[3][3], sa2L[2][3], sa3L[3], fp3L[2], fp2L[2], fp1L[2];
    for (int m = 0; m < 3; ++m) for (int l = 0; l < 3; ++l) nextL(sa1L[m][l]);
    for (int m = 0; m < 2; ++m) for (int l = 0; l < 3; ++l) nextL(sa2L[m][l]);
    for (int l = 0; l < 3; ++l) nextL(sa3L[l]);
    for (int l = 0; l < 2; ++l) nextL(fp3L[l]);
    for (int l = 0; l < 2; ++l) nextL(fp2L[l]);
    for (int l = 0; l < 2; ++l) nextL(fp1L[l]);
    Lyr headL1; nextL(headL1);
    const float* hW2 = F(t); const float* hb2 = F(t + 1);

    // ---- workspace bump allocator
    unsigned char* wp = (unsigned char*)d_ws;
    auto alloc = [&](size_t bytes) {
        void* p = wp; wp += (bytes + 255) & ~(size_t)255; return p;
    };
    float* xt      = (float*)alloc((size_t)B * N * 6 * 4);
    float* coords  = (float*)alloc((size_t)B * N * 3 * 4);
    int*   fidx1   = (int*)  alloc((size_t)B * 512 * 4);
    float* l1_xyz  = (float*)alloc((size_t)B * 512 * 3 * 4);
    int*   fidx2   = (int*)  alloc((size_t)B * 128 * 4);
    float* l2_xyz  = (float*)alloc((size_t)B * 128 * 3 * 4);
    int*   gi      = (int*)  alloc((size_t)B * 512 * 128 * 4);
    float* l1_pts  = (float*)alloc((size_t)B * 512 * 320 * 4);
    float* l2_pts  = (float*)alloc((size_t)B * 128 * 512 * 4);
    float* l3      = (float*)alloc((size_t)B * 1024 * 4);
    int*   knnI    = (int*)  alloc((size_t)B * 2048 * 3 * 4);
    float* knnW    = (float*)alloc((size_t)B * 2048 * 3 * 4);
    const size_t BUFH = (size_t)32768 * 160;             // halfs per ping-pong buffer
    _Float16* bufA16 = (_Float16*)alloc(BUFH * 2);
    _Float16* bufB16 = (_Float16*)alloc(BUFH * 2);

    // ---- f16 weight staging: [pad32(Cout), pad32(Cin)], zero padded
    struct WInfo { const float* W; int Cout, Cin; _Float16* W16; };
    WInfo wi[26]; int nw = 0;
    const int sa1C[3][3] = {{32, 32, 64}, {64, 64, 128}, {64, 96, 128}};
    const int sa2C[2][3] = {{128, 128, 256}, {128, 196, 256}};
    for (int m = 0; m < 3; ++m) {
        int cin = 9;
        for (int l = 0; l < 3; ++l) { wi[nw++] = {sa1L[m][l].W, sa1C[m][l], cin, nullptr}; cin = sa1C[m][l]; }
    }
    for (int m = 0; m < 2; ++m) {
        int cin = 323;
        for (int l = 0; l < 3; ++l) { wi[nw++] = {sa2L[m][l].W, sa2C[m][l], cin, nullptr}; cin = sa2C[m][l]; }
    }
    { int cs[4] = {515, 256, 512, 1024}; for (int l = 0; l < 3; ++l) wi[nw++] = {sa3L[l].W, cs[l + 1], cs[l], nullptr}; }
    wi[nw++] = {fp3L[0].W, 256, 1536, nullptr}; wi[nw++] = {fp3L[1].W, 256, 256, nullptr};
    wi[nw++] = {fp2L[0].W, 256, 576, nullptr};  wi[nw++] = {fp2L[1].W, 128, 256, nullptr};
    wi[nw++] = {fp1L[0].W, 128, 153, nullptr};  wi[nw++] = {fp1L[1].W, 128, 128, nullptr};
    wi[nw++] = {headL1.W, 128, 128, nullptr};   wi[nw++] = {hW2, 50, 128, nullptr};
    for (int i = 0; i < nw; ++i) {
        int coP = pad32(wi[i].Cout); if (coP < 64) coP = 64;
        int ciP = pad32(wi[i].Cin);
        wi[i].W16 = (_Float16*)alloc((size_t)coP * ciP * 2);
        int total = coP * ciP;
        wcvt_kernel<<<ceil_div(total, 256), 256, 0, stream>>>(
            wi[i].W, wi[i].W16, wi[i].Cout, wi[i].Cin, ciP, total);
    }
    // weight index helpers
    auto W16 = [&](int i) { return wi[i].W16; };
    // indices: sa1 m*3+l -> 0..8 ; sa2 9+m*3+l ; sa3 15..17 ; fp3 18,19 ;
    //          fp2 20,21 ; fp1 22,23 ; head 24,25

    auto elems = [&](int total) { return ceil_div(total, 256); };

    // ---- input transpose
    transpose_in_kernel<<<elems(B * N * 6), 256, 0, stream>>>(xyz, xt, coords, B, 6, N);

    // =================== SA1 (npoint=512, MSG x3) ===================
    fps_kernel<<<B, 256, (2048 + 2 * 256) * 4, stream>>>(coords, 2048, 512, fidx1);
    gather3_kernel<<<elems(B * 512 * 3), 256, 0, stream>>>(coords, fidx1, l1_xyz,
                                                           2048, 512, B * 512 * 3);
    {
        const float radii[3] = {0.1f, 0.2f, 0.4f};
        const int   nsamp[3] = {32, 64, 128};
        const int   offs[3]  = {0, 64, 192};
        for (int br = 0; br < 3; ++br) {
            query_ball_kernel<<<elems(B * 512), 256, 0, stream>>>(
                coords, l1_xyz, radii[br] * radii[br], nsamp[br], 2048, 512, gi, B * 512);
            SAParams p{};
            p.points = xt; p.coords = coords; p.new_xyz = l1_xyz; p.gi = gi;
            p.S = 512; p.P = 2048; p.Cpts = 6; p.ns = nsamp[br];
            p.Cin1P = pad32(9);
            const Lyr* L = sa1L[br];
            p.W1 = W16(br * 3 + 0); p.b1 = L[0].b; p.g1 = L[0].g; p.be1 = L[0].be; p.C1 = sa1C[br][0];
            p.W2 = W16(br * 3 + 1); p.b2 = L[1].b; p.g2 = L[1].g; p.be2 = L[1].be; p.C2 = sa1C[br][1];
            p.C2P = pad32(p.C2);
            p.W3 = W16(br * 3 + 2); p.b3 = L[2].b; p.g3 = L[2].g; p.be3 = L[2].be; p.C3 = sa1C[br][2];
            int bh = p.ns * p.Cin1P; int bh2 = p.ns * p.C2P;
            p.actBhalfs = bh > bh2 ? bh : bh2;
            p.out = l1_pts; p.outC = 320; p.outOff = offs[br];
            size_t lds = ((size_t)p.ns * p.C1 + p.actBhalfs) * 2 + (size_t)p.C3 * 4 + (size_t)p.ns * 4;
            sa_fused_kernel<<<B * 512, 256, lds, stream>>>(p);
        }
    }

    // =================== SA2 (npoint=128, MSG x2) ===================
    fps_kernel<<<B, 256, (512 + 2 * 256) * 4, stream>>>(l1_xyz, 512, 128, fidx2);
    gather3_kernel<<<elems(B * 128 * 3), 256, 0, stream>>>(l1_xyz, fidx2, l2_xyz,
                                                           512, 128, B * 128 * 3);
    {
        const float radii[2] = {0.4f, 0.8f};
        const int   nsamp[2] = {64, 128};
        const int   offs[2]  = {0, 256};
        for (int br = 0; br < 2; ++br) {
            query_ball_kernel<<<elems(B * 128), 256, 0, stream>>>(
                l1_xyz, l2_xyz, radii[br] * radii[br], nsamp[br], 512, 128, gi, B * 128);
            SAParams p{};
            p.points = l1_pts; p.coords = l1_xyz; p.new_xyz = l2_xyz; p.gi = gi;
            p.S = 128; p.P = 512; p.Cpts = 320; p.ns = nsamp[br];
            p.Cin1P = pad32(323);
            const Lyr* L = sa2L[br];
            p.W1 = W16(9 + br * 3 + 0); p.b1 = L[0].b; p.g1 = L[0].g; p.be1 = L[0].be; p.C1 = sa2C[br][0];
            p.W2 = W16(9 + br * 3 + 1); p.b2 = L[1].b; p.g2 = L[1].g; p.be2 = L[1].be; p.C2 = sa2C[br][1];
            p.C2P = pad32(p.C2);
            p.W3 = W16(9 + br * 3 + 2); p.b3 = L[2].b; p.g3 = L[2].g; p.be3 = L[2].be; p.C3 = sa2C[br][2];
            int bh = p.ns * p.Cin1P; int bh2 = p.ns * p.C2P;
            p.actBhalfs = bh > bh2 ? bh : bh2;
            p.out = l2_pts; p.outC = 512; p.outOff = offs[br];
            size_t lds = ((size_t)p.ns * p.C1 + p.actBhalfs) * 2 + (size_t)p.C3 * 4 + (size_t)p.ns * 4;
            sa_fused_kernel<<<B * 128, 256, lds, stream>>>(p);
        }
    }

    // =================== SA3 (dense f16 WMMA chain) ===================
    const int M3 = B * 128;
    sa3_build_kernel<<<elems(M3 * 544), 256, 0, stream>>>(l2_xyz, l2_pts, bufA16, M3 * 544);
    launch_linear16(stream, bufA16, W16(15), sa3L[0], bufB16, M3, 515, 256, 3);
    launch_linear16(stream, bufB16, W16(16), sa3L[1], bufA16, M3, 256, 512, 3);
    launch_linear16(stream, bufA16, W16(17), sa3L[2], bufB16, M3, 512, 1024, 3);
    max_over_points16_kernel<<<elems(B * 1024), 256, 0, stream>>>(
        bufB16, l3, 128, 1024, 1024, B * 1024);

    // =================== FP3 ===================
    fp3_build_kernel<<<elems(M3 * 1536), 256, 0, stream>>>(l2_pts, l3, bufA16, M3 * 1536);
    launch_linear16(stream, bufA16, W16(18), fp3L[0], bufB16, M3, 1536, 256, 3);
    launch_linear16(stream, bufB16, W16(19), fp3L[1], bufA16, M3, 256, 256, 3);  // l2_new in A

    // =================== FP2 ===================
    const int M2 = B * 512;
    knn3_kernel<<<elems(M2), 256, 0, stream>>>(l1_xyz, l2_xyz, 512, 128, knnI, knnW, M2);
    fp2_build_kernel<<<elems(M2 * 576), 256, 0, stream>>>(l1_pts, bufA16, knnI, knnW,
                                                          bufB16, M2 * 576);
    launch_linear16(stream, bufB16, W16(20), fp2L[0], bufA16, M2, 576, 256, 3);
    launch_linear16(stream, bufA16, W16(21), fp2L[1], bufB16, M2, 256, 128, 3);  // l1_new in B

    // =================== FP1 ===================
    const int M1 = B * 2048;
    knn3_kernel<<<elems(M1), 256, 0, stream>>>(coords, l1_xyz, 2048, 512, knnI, knnW, M1);
    fp1_build_kernel<<<elems(M1 * 160), 256, 0, stream>>>(cls, coords, xt, bufB16,
                                                          knnI, knnW, bufA16, M1 * 160);
    launch_linear16(stream, bufA16, W16(22), fp1L[0], bufB16, M1, 153, 128, 3);
    launch_linear16(stream, bufB16, W16(23), fp1L[1], bufA16, M1, 128, 128, 3);  // l0 in A

    // =================== Head + log-softmax ===================
    launch_linear16(stream, bufA16, W16(24), headL1, bufB16, M1, 128, 128, 3);
    Lyr head2{hW2, hb2, nullptr, nullptr};
    launch_linear16(stream, bufB16, W16(25), head2, bufA16, M1, 128, 50, 0);  // [M1,64]
    logsoftmax50_kernel<<<elems(M1), 256, 0, stream>>>(bufA16, (float*)d_out, M1);

    // l3_pts output tail [B,1,1024]
    copy_kernel<<<elems(B * 1024), 256, 0, stream>>>(
        l3, (float*)d_out + (size_t)M1 * 50, B * 1024);
}
